// FlashBTLMAttention_24120536334666
// MI455X (gfx1250) — compile-verified, matrix operand order checked by reading, not run on hardware
//
#include <hip/hip_runtime.h>
#include <hip/hip_bf16.h>

// ---------------- problem constants ----------------
#define HIDDEN  2048
#define NQKV    6144            // 3 * HIDDEN
#define NHEADS  16
#define HEADD   128
#define BATCH   4
#define SEQ     2048
#define MROWS   (BATCH * SEQ)   // 8192

// ---------------- tiling constants ----------------
constexpr int BT   = 128;  // block tile (M and N)
constexpr int KT   = 32;   // K step per WMMA
constexpr int LDAU = 20;   // LDS row stride in uints (= 40 bf16 = 80 bytes, 16B aligned, conflict-free)
constexpr int KC   = 32;   // attention key-chunk

// ---------------- vector types ----------------
typedef __attribute__((ext_vector_type(16))) __bf16       v16bf;
typedef __attribute__((ext_vector_type(8)))  float        v8f;
typedef __attribute__((ext_vector_type(4)))  unsigned int u32x4;
typedef __attribute__((ext_vector_type(4)))  float        f32x4;

union FragU { u32x4 u[2]; v16bf b; };

// ---------------- bf16 helpers (round-to-nearest-even) ----------------
__device__ __forceinline__ unsigned pack2_bf16(float a, float b) {
    unsigned ua = __float_as_uint(a), ub = __float_as_uint(b);
    ua = (ua + 0x7FFFu + ((ua >> 16) & 1u)) >> 16;
    ub = (ub + 0x7FFFu + ((ub >> 16) & 1u)) >> 16;
    // v_perm_b32: result = {b_hi16, a_hi16} -- but after shifts a simple or:
    return (ua & 0xFFFFu) | (ub << 16);
}
__device__ __forceinline__ unsigned short f2bf(float a) {
    unsigned ua = __float_as_uint(a);
    ua = (ua + 0x7FFFu + ((ua >> 16) & 1u)) >> 16;
    return (unsigned short)ua;
}

// WMMA wrapper: D = A(16x32 bf16) * B(32x16 bf16) + C(f32)
__device__ __forceinline__ v8f wmma_bf16(v16bf a, v16bf b, v8f c) {
    return __builtin_amdgcn_wmma_f32_16x16x32_bf16(
        /*neg_a=*/false, a, /*neg_b=*/false, b,
        /*c_mod=*/(short)0, c, /*reuse_a=*/false, /*reuse_b=*/false);
}

// ---------------- CDNA5 async global->LDS copy (ASYNCcnt path) ----------------
// GLOBAL_LOAD_ASYNC_TO_LDS_B128: vdst = VGPR holding the LDS byte address
// (low 32 bits of a generic LDS pointer per the aperture rules), vaddr = 64-bit
// global address. Tracked by ASYNCcnt; wave waits s_wait_asynccnt 0 before the
// workgroup barrier so all tiles are visible to consumer waves.
__device__ __forceinline__ void async_b128(const void* lds_ptr, const void* gptr) {
    unsigned loff = (unsigned)(size_t)lds_ptr;
    unsigned long long ga = (unsigned long long)(size_t)gptr;
    asm volatile("global_load_async_to_lds_b128 %0, %1, off"
                 :: "v"(loff), "v"(ga) : "memory");
}
__device__ __forceinline__ void async_wait0() {
    asm volatile("s_wait_asynccnt 0" ::: "memory");
}

// Gather one 16x32 bf16 fragment (A or B operand) from an LDS tile stored
// [row][40 bf16]. Per ISA layout: lane L -> row = L&15, sel = L>>4 picks the
// two contiguous K-octets {sel*8..sel*8+7, 16+sel*8..23+sel*8}.
__device__ __forceinline__ v16bf ld_frag(const unsigned* rowbase, int sel) {
    FragU f;
    f.u[0] = *(const u32x4*)(rowbase + sel * 4);
    f.u[1] = *(const u32x4*)(rowbase + 8 + sel * 4);
    return f.b;
}

// Stage a 128x32 bf16 tile into LDS via async copies. 256 threads, 32B each.
__device__ __forceinline__ void stage_bf16_async(unsigned* lds, const unsigned short* src,
                                                 int ld_src, int tid, int row0, int k0) {
    int r = tid >> 1;            // 0..127
    int c = (tid & 1) * 16;      // 0 or 16
    const unsigned short* g = src + (size_t)(row0 + r) * ld_src + k0 + c;
    unsigned* d = lds + r * LDAU + (c >> 1);
    async_b128(d,     g);
    async_b128(d + 4, g + 8);
}

// ============================================================================
// Kernel 0: one-shot fp32 -> bf16 conversion (bandwidth bound; runs once per
// tensor instead of per GEMM tile, removing all conversion VALU from hot loops)
// ============================================================================
__global__ __launch_bounds__(256)
void f32_to_bf16_kernel(const float* __restrict__ src,
                        unsigned short* __restrict__ dst, int n8) {
    int i = blockIdx.x * 256 + threadIdx.x;   // one thread = 8 elements
    if (i >= n8) return;
    const f32x4* s = (const f32x4*)src + (size_t)i * 2;
    f32x4 x0 = s[0], x1 = s[1];
    u32x4 p = { pack2_bf16(x0.x, x0.y), pack2_bf16(x0.z, x0.w),
                pack2_bf16(x1.x, x1.y), pack2_bf16(x1.z, x1.w) };
    *((u32x4*)dst + i) = p;
}

// ============================================================================
// Kernel 1: QKV = hidden_bf16 @ w_qkv_bf16^T + b_qkv, scattered to Q/K/V
// bf16 [b][h][s][d].  128x128 block tile, 8 waves x (32x64), K-step 32.
// ============================================================================
__global__ __launch_bounds__(256)
void qkv_gemm_kernel(const unsigned short* __restrict__ hidden,
                     const unsigned short* __restrict__ w_qkv,
                     const float* __restrict__ b_qkv,
                     unsigned short* __restrict__ qb,
                     unsigned short* __restrict__ kb,
                     unsigned short* __restrict__ vb) {
    __shared__ __align__(16) unsigned lA[BT * LDAU];
    __shared__ __align__(16) unsigned lB[BT * LDAU];

    const int tid  = threadIdx.x;
    const int bm   = blockIdx.y * BT;
    const int bn   = blockIdx.x * BT;
    const int w    = tid >> 5, lane = tid & 31;
    const int lm   = lane & 15, sel = lane >> 4;
    const int wm   = (w >> 1) * 32;
    const int wn   = (w & 1) * 64;

    v8f acc[2][4];
#pragma unroll
    for (int i = 0; i < 2; ++i)
#pragma unroll
        for (int j = 0; j < 4; ++j) acc[i][j] = (v8f)0.0f;

    for (int k0 = 0; k0 < HIDDEN; k0 += KT) {
        stage_bf16_async(lA, hidden, HIDDEN, tid, bm, k0);
        stage_bf16_async(lB, w_qkv,  HIDDEN, tid, bn, k0);
        async_wait0();
        __syncthreads();

        v16bf af[2], bfr[4];
#pragma unroll
        for (int i = 0; i < 2; ++i)
            af[i] = ld_frag(lA + (wm + i * 16 + lm) * LDAU, sel);
#pragma unroll
        for (int j = 0; j < 4; ++j)
            bfr[j] = ld_frag(lB + (wn + j * 16 + lm) * LDAU, sel);
#pragma unroll
        for (int i = 0; i < 2; ++i)
#pragma unroll
            for (int j = 0; j < 4; ++j)
                acc[i][j] = wmma_bf16(af[i], bfr[j], acc[i][j]);
        __syncthreads();
    }

    // Epilogue: bias + scatter bf16 into Q/K/V [b][h][s][d]
#pragma unroll
    for (int j = 0; j < 4; ++j) {
        int ng = bn + wn + j * 16 + lm;          // global output column (0..6143)
        float bias = b_qkv[ng];
        int part = ng >> 11;                      // 0=Q 1=K 2=V
        int rem  = ng & 2047;
        int head = rem >> 7, d = rem & 127;
        unsigned short* dst = (part == 0) ? qb : ((part == 1) ? kb : vb);
#pragma unroll
        for (int i = 0; i < 2; ++i) {
            int mbase = bm + wm + i * 16 + sel * 8;
#pragma unroll
            for (int r = 0; r < 8; ++r) {
                int mg = mbase + r;
                int bb = mg >> 11, s = mg & 2047;
                size_t off = (((size_t)(bb * NHEADS + head)) * SEQ + s) * HEADD + d;
                dst[off] = f2bf(acc[i][j][r] + bias);
            }
        }
    }
}

// ============================================================================
// Kernel 2: causal flash attention, bf16 WMMA, fp32 online softmax.
// Grid: (S/128, NHEADS, BATCH); block 256 = 8 waves, each owns 16 q-rows.
// ============================================================================
__global__ __launch_bounds__(256)
void flash_attn_kernel(const unsigned short* __restrict__ qb,
                       const unsigned short* __restrict__ kb,
                       const unsigned short* __restrict__ vb,
                       unsigned short* __restrict__ attn) {
    __shared__ __align__(16) unsigned       lK[KC * 68];        // [32 keys][136 bf16] padded
    __shared__ __align__(16) unsigned short lV[HEADD * 40];     // transposed: [d][40] padded
    __shared__ __align__(16) unsigned short lP[8][16 * 40];     // per-wave P tile [q][40]

    const int tid = threadIdx.x, w = tid >> 5, lane = tid & 31;
    const int lm = lane & 15, sel = lane >> 4;
    const int bb = blockIdx.z, h = blockIdx.y, qblk = blockIdx.x;
    const int qr = qblk * 128 + w * 16;   // this wave's first q row

    const size_t headoff = ((size_t)(bb * NHEADS + h)) * SEQ * HEADD;
    const unsigned short* Q = qb + headoff;
    const unsigned short* K = kb + headoff;
    const unsigned short* V = vb + headoff;

    // Q fragments: 4 K-slices of 32 along d; rows contiguous in d.
    v16bf qf[4];
    {
        const unsigned short* qrow = Q + (size_t)(qr + lm) * HEADD;
#pragma unroll
        for (int di = 0; di < 4; ++di) {
            FragU f;
            f.u[0] = *(const u32x4*)(qrow + di * 32 + sel * 8);
            f.u[1] = *(const u32x4*)(qrow + di * 32 + 16 + sel * 8);
            qf[di] = f.b;
        }
    }

    v8f o[8];
#pragma unroll
    for (int dt = 0; dt < 8; ++dt) o[dt] = (v8f)0.0f;
    float mi[8], li[8];
#pragma unroll
    for (int r = 0; r < 8; ++r) { mi[r] = -3.0e38f; li[r] = 0.0f; }

    const float scale = 0.08838834764831844f;   // 1/sqrt(128)
    const int nchunks = (qblk + 1) * (128 / KC);

    for (int c = 0; c < nchunks; ++c) {
        const int j0 = c * KC;
        // ---- stage K chunk (async row-major copy) and V chunk (transposed) ----
        {
            int r  = tid >> 3;             // 0..31  key row
            int cc = (tid & 7) * 16;       // 0..112 d column (ushort units)
            const unsigned short* ks = K + (size_t)(j0 + r) * HEADD + cc;
            unsigned* dk = lK + r * 68 + (cc >> 1);
            async_b128(dk,     ks);
            async_b128(dk + 4, ks + 8);
            const unsigned short* vs = V + (size_t)(j0 + r) * HEADD + cc;
            u32x4 vv0 = *(const u32x4*)(vs);
            u32x4 vv1 = *(const u32x4*)(vs + 8);
            const unsigned short* ve = (const unsigned short*)&vv0;
#pragma unroll
            for (int i = 0; i < 8; ++i)
                lV[(cc + i) * 40 + r] = ve[i];
            const unsigned short* vo = (const unsigned short*)&vv1;
#pragma unroll
            for (int i = 0; i < 8; ++i)
                lV[(cc + 8 + i) * 40 + r] = vo[i];
        }
        async_wait0();
        __syncthreads();

        if (j0 <= qr + 15) {               // causally relevant for this wave
            // ---- S = Q K^T (2 key sub-tiles of 16, accumulate over d) ----
            v8f s0 = (v8f)0.0f, s1 = (v8f)0.0f;
#pragma unroll
            for (int di = 0; di < 4; ++di) {
                FragU fk0, fk1;
                const unsigned* kr0 = lK + (0 * 16 + lm) * 68 + di * 16;
                const unsigned* kr1 = lK + (1 * 16 + lm) * 68 + di * 16;
                fk0.u[0] = *(const u32x4*)(kr0 + sel * 4);
                fk0.u[1] = *(const u32x4*)(kr0 + 8 + sel * 4);
                fk1.u[0] = *(const u32x4*)(kr1 + sel * 4);
                fk1.u[1] = *(const u32x4*)(kr1 + 8 + sel * 4);
                s0 = wmma_bf16(qf[di], fk0.b, s0);
                s1 = wmma_bf16(qf[di], fk1.b, s1);
            }
            // ---- scale + causal mask ----
            const int qgl = qr + sel * 8;
            const int jg0 = j0 + lm, jg1 = j0 + 16 + lm;
#pragma unroll
            for (int r = 0; r < 8; ++r) {
                int qg = qgl + r;
                float v0 = s0[r] * scale, v1 = s1[r] * scale;
                if (jg0 > qg) v0 = -3.0e38f;
                if (jg1 > qg) v1 = -3.0e38f;
                s0[r] = v0; s1[r] = v1;
            }
            // ---- online softmax per row (cross-lane reduce within 16-lane groups) ----
            unsigned short* pbuf = lP[w];
#pragma unroll
            for (int r = 0; r < 8; ++r) {
                float mx = fmaxf(s0[r], s1[r]);
                mx = fmaxf(mx, __shfl_xor(mx, 1, 32));
                mx = fmaxf(mx, __shfl_xor(mx, 2, 32));
                mx = fmaxf(mx, __shfl_xor(mx, 4, 32));
                mx = fmaxf(mx, __shfl_xor(mx, 8, 32));
                float mnew  = fmaxf(mi[r], mx);
                float alpha = __expf(mi[r] - mnew);
                float p0 = __expf(s0[r] - mnew);
                float p1 = __expf(s1[r] - mnew);
                float ps = p0 + p1;
                ps += __shfl_xor(ps, 1, 32);
                ps += __shfl_xor(ps, 2, 32);
                ps += __shfl_xor(ps, 4, 32);
                ps += __shfl_xor(ps, 8, 32);
                li[r] = li[r] * alpha + ps;
                mi[r] = mnew;
#pragma unroll
                for (int dt = 0; dt < 8; ++dt) o[dt][r] *= alpha;
                int ql = r + sel * 8;
                pbuf[ql * 40 + lm]      = f2bf(p0);
                pbuf[ql * 40 + 16 + lm] = f2bf(p1);
            }
            // ---- O += P V ----
            FragU fp;
            const unsigned short* pr = pbuf + lm * 40;
            fp.u[0] = *(const u32x4*)(pr + sel * 8);
            fp.u[1] = *(const u32x4*)(pr + 16 + sel * 8);
            v16bf pf = fp.b;
#pragma unroll
            for (int dt = 0; dt < 8; ++dt) {
                FragU fv;
                const unsigned short* vr = lV + (dt * 16 + lm) * 40;
                fv.u[0] = *(const u32x4*)(vr + sel * 8);
                fv.u[1] = *(const u32x4*)(vr + 16 + sel * 8);
                o[dt] = wmma_bf16(pf, fv.b, o[dt]);
            }
        }
        __syncthreads();
    }

    // ---- normalize and write bf16 attn output [b*S+s][h*128+d] ----
#pragma unroll
    for (int r = 0; r < 8; ++r) {
        float inv = 1.0f / li[r];
        int qg = qr + r + sel * 8;
        size_t rowoff = ((size_t)(bb * SEQ + qg)) * HIDDEN + h * HEADD;
#pragma unroll
        for (int dt = 0; dt < 8; ++dt)
            attn[rowoff + dt * 16 + lm] = f2bf(o[dt][r] * inv);
    }
}

// ============================================================================
// Kernel 3: out = attn(bf16) @ w_dense_bf16^T + b_dense  (fp32 output)
// ============================================================================
__global__ __launch_bounds__(256)
void dense_gemm_kernel(const unsigned short* __restrict__ A,
                       const unsigned short* __restrict__ w_dense,
                       const float* __restrict__ b_dense,
                       float* __restrict__ out) {
    __shared__ __align__(16) unsigned lA[BT * LDAU];
    __shared__ __align__(16) unsigned lB[BT * LDAU];

    const int tid = threadIdx.x;
    const int bm  = blockIdx.y * BT;
    const int bn  = blockIdx.x * BT;
    const int w   = tid >> 5, lane = tid & 31;
    const int lm  = lane & 15, sel = lane >> 4;
    const int wm  = (w >> 1) * 32;
    const int wn  = (w & 1) * 64;

    v8f acc[2][4];
#pragma unroll
    for (int i = 0; i < 2; ++i)
#pragma unroll
        for (int j = 0; j < 4; ++j) acc[i][j] = (v8f)0.0f;

    for (int k0 = 0; k0 < HIDDEN; k0 += KT) {
        stage_bf16_async(lA, A,       HIDDEN, tid, bm, k0);
        stage_bf16_async(lB, w_dense, HIDDEN, tid, bn, k0);
        async_wait0();
        __syncthreads();

        v16bf af[2], bfr[4];
#pragma unroll
        for (int i = 0; i < 2; ++i)
            af[i] = ld_frag(lA + (wm + i * 16 + lm) * LDAU, sel);
#pragma unroll
        for (int j = 0; j < 4; ++j)
            bfr[j] = ld_frag(lB + (wn + j * 16 + lm) * LDAU, sel);
#pragma unroll
        for (int i = 0; i < 2; ++i)
#pragma unroll
            for (int j = 0; j < 4; ++j)
                acc[i][j] = wmma_bf16(af[i], bfr[j], acc[i][j]);
        __syncthreads();
    }

#pragma unroll
    for (int j = 0; j < 4; ++j) {
        int ng = bn + wn + j * 16 + lm;
        float bias = b_dense[ng];
#pragma unroll
        for (int i = 0; i < 2; ++i) {
            int mbase = bm + wm + i * 16 + sel * 8;
#pragma unroll
            for (int r = 0; r < 8; ++r)
                out[(size_t)(mbase + r) * HIDDEN + ng] = acc[i][j][r] + bias;
        }
    }
}

// ============================================================================
extern "C" void kernel_launch(void* const* d_in, const int* in_sizes, int n_in,
                              void* d_out, int out_size, void* d_ws, size_t ws_size,
                              hipStream_t stream) {
    (void)in_sizes; (void)n_in; (void)out_size; (void)ws_size;
    const float* hidden  = (const float*)d_in[0];
    const float* w_qkv   = (const float*)d_in[1];
    const float* b_qkv   = (const float*)d_in[2];
    const float* w_dense = (const float*)d_in[3];
    const float* b_dense = (const float*)d_in[4];

    unsigned short* ws = (unsigned short*)d_ws;
    const size_t HB  = (size_t)MROWS * HIDDEN;    // 16,777,216
    const size_t WQB = (size_t)NQKV * HIDDEN;     // 12,582,912
    const size_t WDB = (size_t)HIDDEN * HIDDEN;   //  4,194,304
    const size_t QE  = (size_t)BATCH * NHEADS * SEQ * HEADD;  // 16,777,216

    unsigned short* hb   = ws;                 // bf16 hidden_states
    unsigned short* wqb  = hb  + HB;           // bf16 w_qkv
    unsigned short* wdb  = wqb + WQB;          // bf16 w_dense
    unsigned short* qb   = wdb + WDB;
    unsigned short* kb   = qb  + QE;
    unsigned short* vb   = kb  + QE;
    unsigned short* attn = vb  + QE;           // [8192][2048] bf16

    // one-shot fp32 -> bf16 conversions (bandwidth bound)
    f32_to_bf16_kernel<<<(int)(HB  / 8 / 256), 256, 0, stream>>>(hidden,  hb,  (int)(HB  / 8));
    f32_to_bf16_kernel<<<(int)(WQB / 8 / 256), 256, 0, stream>>>(w_qkv,   wqb, (int)(WQB / 8));
    f32_to_bf16_kernel<<<(int)(WDB / 8 / 256), 256, 0, stream>>>(w_dense, wdb, (int)(WDB / 8));

    dim3 g1(NQKV / BT, MROWS / BT);      // 48 x 64
    qkv_gemm_kernel<<<g1, 256, 0, stream>>>(hb, wqb, b_qkv, qb, kb, vb);

    dim3 g2(SEQ / 128, NHEADS, BATCH);   // 16 x 16 x 4
    flash_attn_kernel<<<g2, 256, 0, stream>>>(qb, kb, vb, attn);

    dim3 g3(HIDDEN / BT, MROWS / BT);    // 16 x 64
    dense_gemm_kernel<<<g3, 256, 0, stream>>>(attn, wdb, b_dense, (float*)d_out);
}
